// VGAE_41540923687532
// MI455X (gfx1250) — matrix operand order repaired and checked by reading, not compiled, not run on hardware
//
#include <hip/hip_runtime.h>

#define LRELU_ALPHA 0.2f

typedef __attribute__((ext_vector_type(16))) __bf16 v16bf;
typedef __attribute__((ext_vector_type(8)))  __bf16 v8bf;
typedef __attribute__((ext_vector_type(8)))  float  v8f;
typedef __attribute__((ext_vector_type(4)))  float  v4f;
typedef __attribute__((ext_vector_type(4)))  int    v4i;

union ATile { v16bf v; v8bf h[2]; __bf16 e[16]; };

__device__ __forceinline__ __bf16 f2bf(float f) {
  union { float f; unsigned u; } v; v.f = f;
  unsigned r = v.u + 0x7FFFu + ((v.u >> 16) & 1u);
  unsigned short hbits = (unsigned short)(r >> 16);
  return __builtin_bit_cast(__bf16, hbits);
}

// ---------------------------------------------------------------- elementwise f32 -> bf16
__global__ void k_cvt_bf16(const float* __restrict__ in, __bf16* __restrict__ out, int n) {
  int i = blockIdx.x * blockDim.x + threadIdx.x;
  if (i < n) out[i] = f2bf(in[i]);
}

// ---------------------------------------------------------------- transpose f32[R][C] -> bf16[C][R]
__global__ void k_transpose_bf16(const float* __restrict__ in, __bf16* __restrict__ out,
                                 int R, int C) {
  __shared__ float tile[32][33];
  int c0 = blockIdx.x * 32, r0 = blockIdx.y * 32;
  for (int dy = threadIdx.y; dy < 32; dy += 8) {
    int r = r0 + dy, c = c0 + (int)threadIdx.x;
    tile[dy][threadIdx.x] = (r < R && c < C) ? in[(size_t)r * C + c] : 0.f;
  }
  __syncthreads();
  for (int dy = threadIdx.y; dy < 32; dy += 8) {
    int c = c0 + dy, r = r0 + (int)threadIdx.x;
    if (c < C && r < R) out[(size_t)c * R + r] = f2bf(tile[threadIdx.x][dy]);
  }
}

// ---------------------------------------------------------------- C_f32[M,Nn] = A_bf16[M,K] @ BT_bf16[Nn,K]^T
__global__ void __launch_bounds__(32) k_gemm_bf16(const __bf16* __restrict__ A,
                                                  const __bf16* __restrict__ BT,
                                                  float* __restrict__ C,
                                                  int M, int Nn, int K) {
  int lane = threadIdx.x;
  int lrow = lane & 15;
  int hi   = (lane >> 4) & 1;
  int mbase = blockIdx.y << 4, nbase = blockIdx.x << 4;
  const __bf16* arow = A  + (size_t)(mbase + lrow) * K + (hi ? 16 : 0);
  const __bf16* brow = BT + (size_t)(nbase + lrow) * K + (hi ? 16 : 0);
  v8f acc;
#pragma unroll
  for (int i = 0; i < 8; ++i) acc[i] = 0.f;
  for (int k = 0; k < K; k += 32) {
    ATile a;
    a.h[0] = *(const v8bf*)(arow + k);
    a.h[1] = *(const v8bf*)(arow + k + 8);
    v16bf b = *(const v16bf*)(brow + k);
    acc = __builtin_amdgcn_wmma_f32_16x16x32_bf16(false, a.v, false, b, (short)0, acc,
                                                  false, false);
  }
  float* cp = C + (size_t)(mbase + (hi ? 8 : 0)) * Nn + nbase + lrow;
#pragma unroll
  for (int r = 0; r < 8; ++r) cp[(size_t)r * Nn] = acc[r];
}

// ---------------------------------------------------------------- s1[i]=Wh[i,:]·a[0:F], s2[i]=Wh[i,:]·a[F:2F]
__global__ void k_gemv2(const float* __restrict__ Wh, const float* __restrict__ a,
                        float* __restrict__ s1, float* __restrict__ s2, int N, int F) {
  int wave = (int)((blockIdx.x * blockDim.x + threadIdx.x) >> 5);
  int lane = threadIdx.x & 31;
  if (wave >= N) return;
  const float* row = Wh + (size_t)wave * F;
  float acc1 = 0.f, acc2 = 0.f;
  for (int f = lane; f < F; f += 32) {
    float w = row[f];
    acc1 += w * a[f];
    acc2 += w * a[F + f];
  }
  for (int off = 16; off; off >>= 1) {
    acc1 += __shfl_down(acc1, off, 32);
    acc2 += __shfl_down(acc2, off, 32);
  }
  if (lane == 0) { s1[wave] = acc1; s2[wave] = acc2; }
}

// ---------------------------------------------------------------- single-block max reduce
__global__ void k_max(const float* __restrict__ in, float* __restrict__ out, int n) {
  __shared__ float sm[1024];
  float m = -3.4e38f;
  for (int i = threadIdx.x; i < n; i += blockDim.x) m = fmaxf(m, in[i]);
  sm[threadIdx.x] = m;
  __syncthreads();
  for (int s = blockDim.x >> 1; s; s >>= 1) {
    if ((int)threadIdx.x < s) sm[threadIdx.x] = fmaxf(sm[threadIdx.x], sm[threadIdx.x + s]);
    __syncthreads();
  }
  if (threadIdx.x == 0) out[0] = sm[0];
}

// ---------------------------------------------------------------- fused GAT attention:
// out[i,:] = act( (sum_j adj_ij * exp(f(s1_i+s2_j)-c_i) * Wh[j,:]) / Z_i )
// one wave owns 16 rows x full F; single pass over adj (shift by c_i = f(s1_i + max s2)).
// EACT: 0 = relu (layer1: relu∘elu == relu), 1 = elu.
template <int NT, int EACT>
__global__ void __launch_bounds__(32) k_gat_attn(const int* __restrict__ adj,
                                                 const __bf16* __restrict__ WhT,
                                                 const float* __restrict__ s1,
                                                 const float* __restrict__ s2,
                                                 const float* __restrict__ Mmax,
                                                 float* __restrict__ out, int Nn) {
  const int F = NT * 16;
  int lane = threadIdx.x;
  int lrow = lane & 15;
  int hi   = (lane >> 4) & 1;
  int rbase = blockIdx.x << 4;
  int myrow = rbase + lrow;

  float s1v = s1[myrow];
  float cm  = s1v + Mmax[0];
  float c   = cm > 0.f ? cm : LRELU_ALPHA * cm;

  const int klo = hi ? 8 : 0;
  const int*   adjrow = adj + (size_t)myrow * Nn + klo;
  const float* s2p    = s2 + klo;
  const __bf16* bbase = WhT + (size_t)lrow * Nn + (hi ? 8 : 0);

  v8f acc[NT];
#pragma unroll
  for (int t = 0; t < NT; ++t)
#pragma unroll
    for (int i = 0; i < 8; ++i) acc[t][i] = 0.f;

  float zpart = 0.f;
  for (int j = 0; j < Nn; j += 32) {
    __builtin_prefetch(adjrow + j + 64, 0, 1);
    v4i a0 = *(const v4i*)(adjrow + j);
    v4i a1 = *(const v4i*)(adjrow + j + 4);
    v4i a2 = *(const v4i*)(adjrow + j + 16);
    v4i a3 = *(const v4i*)(adjrow + j + 20);
    v4f q0 = *(const v4f*)(s2p + j);
    v4f q1 = *(const v4f*)(s2p + j + 4);
    v4f q2 = *(const v4f*)(s2p + j + 16);
    v4f q3 = *(const v4f*)(s2p + j + 20);
    int   av[16]; float qv[16];
    *(v4i*)(av + 0) = a0;  *(v4i*)(av + 4)  = a1;
    *(v4i*)(av + 8) = a2;  *(v4i*)(av + 12) = a3;
    *(v4f*)(qv + 0) = q0;  *(v4f*)(qv + 4)  = q1;
    *(v4f*)(qv + 8) = q2;  *(v4f*)(qv + 12) = q3;
    ATile A;
#pragma unroll
    for (int e = 0; e < 16; ++e) {
      float t0 = s1v + qv[e];
      float fe = t0 > 0.f ? t0 : LRELU_ALPHA * t0;
      float pe = av[e] > 0 ? __expf(fe - c) : 0.f;
      zpart += pe;
      A.e[e] = f2bf(pe);
    }
#pragma unroll
    for (int t = 0; t < NT; ++t) {
      ATile bt;
      bt.h[0] = *(const v8bf*)(bbase + (size_t)t * 16 * Nn + j);
      bt.h[1] = *(const v8bf*)(bbase + (size_t)t * 16 * Nn + j + 16);
      v16bf b = bt.v;
      acc[t] = __builtin_amdgcn_wmma_f32_16x16x32_bf16(false, A.v, false, b, (short)0,
                                                       acc[t], false, false);
    }
  }

  // full row denominator: lane L and L^16 hold complementary K halves of row L%16
  float zfull = zpart + __shfl_xor(zpart, 16, 32);
  float zr[8];
#pragma unroll
  for (int r = 0; r < 8; ++r) zr[r] = __shfl(zfull, r + (hi ? 8 : 0), 32);

  float* orow = out + (size_t)(rbase + (hi ? 8 : 0)) * F + lrow;
#pragma unroll
  for (int t = 0; t < NT; ++t)
#pragma unroll
    for (int r = 0; r < 8; ++r) {
      float v = acc[t][r] / zr[r];
      if (EACT == 0) v = fmaxf(v, 0.f);
      else           v = v > 0.f ? v : (__expf(v) - 1.f);
      orow[(size_t)r * F + t * 16] = v;
    }
}

// ---------------------------------------------------------------- z = eps*exp(logstd)+mu (f32 out + bf16 copy)
__global__ void k_z(const float* __restrict__ eps, const float* __restrict__ mu,
                    const float* __restrict__ ls, float* __restrict__ zout,
                    __bf16* __restrict__ zb, int n) {
  int i = blockIdx.x * blockDim.x + threadIdx.x;
  if (i < n) {
    float z = eps[i] * __expf(ls[i]) + mu[i];
    zout[i] = z;
    zb[i]   = f2bf(z);
  }
}

// ---------------------------------------------------------------- adj_hat = sigmoid(z @ z^T), 16x16 tiles
__global__ void __launch_bounds__(32) k_zzt_sigmoid(const __bf16* __restrict__ zb,
                                                    float* __restrict__ out, int Nn, int K) {
  int lane = threadIdx.x;
  int lrow = lane & 15;
  int hi   = (lane >> 4) & 1;
  int mbase = blockIdx.y << 4, nbase = blockIdx.x << 4;
  const __bf16* arow = zb + (size_t)(mbase + lrow) * K + (hi ? 16 : 0);
  const __bf16* brow = zb + (size_t)(nbase + lrow) * K + (hi ? 16 : 0);  // B[k][n] = z[n][k]
  v8f acc;
#pragma unroll
  for (int i = 0; i < 8; ++i) acc[i] = 0.f;
  for (int k = 0; k < K; k += 32) {
    ATile a;
    a.h[0] = *(const v8bf*)(arow + k);
    a.h[1] = *(const v8bf*)(arow + k + 8);
    v16bf b = *(const v16bf*)(brow + k);
    acc = __builtin_amdgcn_wmma_f32_16x16x32_bf16(false, a.v, false, b, (short)0, acc,
                                                  false, false);
  }
  float* cp = out + (size_t)(mbase + (hi ? 8 : 0)) * Nn + nbase + lrow;
#pragma unroll
  for (int r = 0; r < 8; ++r) cp[(size_t)r * Nn] = 1.f / (1.f + __expf(-acc[r]));
}

// ================================================================ host
extern "C" void kernel_launch(void* const* d_in, const int* in_sizes, int n_in,
                              void* d_out, int out_size, void* d_ws, size_t ws_size,
                              hipStream_t stream) {
  (void)in_sizes; (void)n_in; (void)out_size; (void)ws_size;
  const int N = 8192, FIN = 512, H1 = 256, H2 = 128;

  const float* x   = (const float*)d_in[0];
  const int*   adj = (const int*)d_in[1];
  const float* eps = (const float*)d_in[2];
  const float* W1  = (const float*)d_in[3];
  const float* a1  = (const float*)d_in[4];
  const float* W2  = (const float*)d_in[5];
  const float* a2  = (const float*)d_in[6];
  const float* W3  = (const float*)d_in[7];
  const float* a3  = (const float*)d_in[8];

  float* adj_hat = (float*)d_out;
  float* z_out   = adj_hat + (size_t)N * N;
  float* mu_out  = z_out + (size_t)N * H2;
  float* ls_out  = mu_out + (size_t)N * H2;

  // workspace layout (aliased buffers; all uses are strictly ordered in-stream)
  char*  P   = (char*)d_ws;
  size_t off = 0;
  auto take = [&](size_t bytes) -> void* {
    void* p = (void*)(P + off);
    off += (bytes + 255) & ~(size_t)255;
    return p;
  };
  void*   bufA = take((size_t)N * FIN * 2);  // xb (bf16) -> hid1 (f32, same 8MB)
  void*   bufB = take((size_t)N * H1 * 4);   // Wh1 (f32) -> Wh2/Wh3 (f32)
  void*   bufC = take((size_t)H1 * N * 2);   // Wh1T (bf16) -> hid1b (bf16)
  void*   bufD = take((size_t)H2 * N * 2);   // Wh2T/Wh3T (bf16) -> zb (bf16)
  __bf16* W1T  = (__bf16*)take((size_t)FIN * H1 * 2);
  __bf16* W2T  = (__bf16*)take((size_t)H1 * H2 * 2);
  __bf16* W3T  = (__bf16*)take((size_t)H1 * H2 * 2);
  float*  s1   = (float*)take((size_t)N * 4);
  float*  s2   = (float*)take((size_t)N * 4);
  float*  Mmax = (float*)take(256);

  __bf16* xb    = (__bf16*)bufA;
  float*  hid1  = (float*)bufA;
  float*  Wh1   = (float*)bufB;
  float*  Wh23  = (float*)bufB;
  __bf16* Wh1T  = (__bf16*)bufC;
  __bf16* hid1b = (__bf16*)bufC;
  __bf16* Wh23T = (__bf16*)bufD;
  __bf16* zb    = (__bf16*)bufD;

  dim3 tb(32, 8);

  // ---- weights to bf16 (transposed so GEMM B-operand loads are contiguous in K)
  k_transpose_bf16<<<dim3(H1 / 32, FIN / 32), tb, 0, stream>>>(W1, W1T, FIN, H1);
  k_transpose_bf16<<<dim3(H2 / 32, H1 / 32), tb, 0, stream>>>(W2, W2T, H1, H2);
  k_transpose_bf16<<<dim3(H2 / 32, H1 / 32), tb, 0, stream>>>(W3, W3T, H1, H2);

  // ---- layer 1
  k_cvt_bf16<<<(N * FIN + 255) / 256, 256, 0, stream>>>(x, xb, N * FIN);
  k_gemm_bf16<<<dim3(H1 / 16, N / 16), 32, 0, stream>>>(xb, W1T, Wh1, N, H1, FIN);
  k_transpose_bf16<<<dim3(H1 / 32, N / 32), tb, 0, stream>>>(Wh1, Wh1T, N, H1);
  k_gemv2<<<N / 8, 256, 0, stream>>>(Wh1, a1, s1, s2, N, H1);
  k_max<<<1, 1024, 0, stream>>>(s2, Mmax, N);
  k_gat_attn<H1 / 16, 0><<<N / 16, 32, 0, stream>>>(adj, Wh1T, s1, s2, Mmax, hid1, N);
  k_cvt_bf16<<<(N * H1 + 255) / 256, 256, 0, stream>>>(hid1, hid1b, N * H1);

  // ---- layer 2 (mu)
  k_gemm_bf16<<<dim3(H2 / 16, N / 16), 32, 0, stream>>>(hid1b, W2T, Wh23, N, H2, H1);
  k_transpose_bf16<<<dim3(H2 / 32, N / 32), tb, 0, stream>>>(Wh23, Wh23T, N, H2);
  k_gemv2<<<N / 8, 256, 0, stream>>>(Wh23, a2, s1, s2, N, H2);
  k_max<<<1, 1024, 0, stream>>>(s2, Mmax, N);
  k_gat_attn<H2 / 16, 1><<<N / 16, 32, 0, stream>>>(adj, Wh23T, s1, s2, Mmax, mu_out, N);

  // ---- layer 3 (logstd), reuses layer-2 buffers
  k_gemm_bf16<<<dim3(H2 / 16, N / 16), 32, 0, stream>>>(hid1b, W3T, Wh23, N, H2, H1);
  k_transpose_bf16<<<dim3(H2 / 32, N / 32), tb, 0, stream>>>(Wh23, Wh23T, N, H2);
  k_gemv2<<<N / 8, 256, 0, stream>>>(Wh23, a3, s1, s2, N, H2);
  k_max<<<1, 1024, 0, stream>>>(s2, Mmax, N);
  k_gat_attn<H2 / 16, 1><<<N / 16, 32, 0, stream>>>(adj, Wh23T, s1, s2, Mmax, ls_out, N);

  // ---- z and adj_hat
  k_z<<<(N * H2 + 255) / 256, 256, 0, stream>>>(eps, mu_out, ls_out, z_out, zb, N * H2);
  k_zzt_sigmoid<<<dim3(N / 16, N / 16), 32, 0, stream>>>(zb, adj_hat, N, H2);
}
